// ConvEmbedding2_2164663517775
// MI455X (gfx1250) — matrix operand order repaired and checked by reading, not compiled, not run on hardware
//
#include <hip/hip_runtime.h>

// ---------------------------------------------------------------------------
// ConvEmbedding as a bf16-WMMA GEMM on gfx1250 (MI455X)
//   C[65536 x 256] = A[65536 x 2304] * B[2304 x 256] + bias
//   A row m = concat_k emb_bf16[clip(x_m + k - 4)][:]
//       staged into LDS in WMMA A-fragment order via GLOBAL_LOAD_ASYNC_TO_LDS
//   B pre-swizzled into v_wmma_f32_16x16x32_bf16 B-fragment layout in d_ws
// ---------------------------------------------------------------------------

typedef __bf16 v16bf __attribute__((ext_vector_type(16)));
typedef float  v8f   __attribute__((ext_vector_type(8)));

#define INP_SIZE 50257
#define HIDDEN   256
#define KSIZE    9
#define NTOK     65536            // B*T = 16*4096
#define KTOT     2304             // KSIZE*HIDDEN
#define NCHUNK   72               // KTOT / 32
#define TOK_PER_WG 128
#define NWG      (NTOK / TOK_PER_WG)   // 512

// ws layout: [0, 25731584)           : emb in bf16 (50257*256 ushort)
//            [25731584, +1179648)    : pre-swizzled B fragments (72*16*32*16 ushort)
#define WS_EMB_ELEMS   ((size_t)INP_SIZE * HIDDEN)          // 12,865,792
#define WS_B_ELEMS     ((size_t)NCHUNK * 16 * 32 * 16)      //    589,824

__device__ __forceinline__ unsigned short f32_to_bf16_rne(float f) {
    unsigned int u = __float_as_uint(f);
    unsigned int r = (u + 0x7FFFu + ((u >> 16) & 1u)) >> 16;
    return (unsigned short)r;
}

// ---- prep 1: emb f32 -> bf16 table ----------------------------------------
__global__ void prep_emb_bf16(const float* __restrict__ emb,
                              unsigned short* __restrict__ out) {
    const int n4 = (int)(WS_EMB_ELEMS / 4);
    int i4 = blockIdx.x * blockDim.x + threadIdx.x;
    if (i4 >= n4) return;
    float4 v = ((const float4*)emb)[i4];
    unsigned long long w =
        (unsigned long long)f32_to_bf16_rne(v.x)
      | ((unsigned long long)f32_to_bf16_rne(v.y) << 16)
      | ((unsigned long long)f32_to_bf16_rne(v.z) << 32)
      | ((unsigned long long)f32_to_bf16_rne(v.w) << 48);
    ((unsigned long long*)out)[i4] = w;
}

// ---- prep 2: W[o,hi,k] -> bf16 B fragments Bfrag[kk][nt][lane][e] ---------
// B-matrix 32x16 bf16 fragment layout (ISA 7.12.2): lane L holds column
// n = nt*16 + (L&15), element e -> k = (e>>3)*16 + (L>>4)*8 + (e&7).
__global__ void prep_w_swizzle(const float* __restrict__ W,
                               unsigned short* __restrict__ outB) {
    unsigned tid = blockIdx.x * blockDim.x + threadIdx.x;   // < 589824, exact grid
    unsigned e    = tid & 15u;
    unsigned lane = (tid >> 4) & 31u;
    unsigned nt   = (tid >> 9) & 15u;
    unsigned kk   = tid >> 13;                 // 0..71
    unsigned kc   = ((e >> 3) << 4) + ((lane >> 4) << 3) + (e & 7u);    // 0..31 within chunk
    unsigned n    = (nt << 4) + (lane & 15u);  // output channel 0..255
    unsigned tap  = kk >> 3;                   // 0..8
    unsigned hi   = ((kk & 7u) << 5) + kc;     // 0..255
    outB[tid] = f32_to_bf16_rne(W[(size_t)n * KTOT + (size_t)hi * KSIZE + tap]);
}

// ---- main GEMM kernel ------------------------------------------------------
// 256 threads = 8 waves. Each wave: 2 M-tiles (32 tokens) x 8 N-tiles (128 out).
__global__ __launch_bounds__(256)
void conv_emb_wmma(const int*            __restrict__ x,
                   const unsigned short* __restrict__ embbf,
                   const unsigned short* __restrict__ wsB,
                   const float*          __restrict__ bias,
                   float*                __restrict__ out) {
    // A tile (128 tokens x 32 K), double buffered, in exact A-fragment order:
    //   Alds[buf][(mt*32 + lane)*16 + e], mt = 0..7  -> 8 KB per buffer
    __shared__ __align__(32) unsigned short Alds[2][8 * 32 * 16];
    __shared__ int sidx[TOK_PER_WG];

    const unsigned tid = threadIdx.x;
    const unsigned wg  = blockIdx.x;

    if (tid < TOK_PER_WG) sidx[tid] = x[wg * TOK_PER_WG + tid];
    __syncthreads();

    // --- staging identity: each thread owns one 16B segment for token m0 and
    // the matching segment for token m0+64.  A-fragment layout (ISA 7.12.2):
    //   lanes 0-15 : K in {0..7} (e=0..7) and {16..23} (e=8..15)
    //   lanes 16-31: K in {8..15}         and {24..31}
    const unsigned m0 = tid >> 2;          // token 0..63 (second token = m0+64)
    const unsigned p  = tid & 3u;
    const int xm0 = sidx[m0];
    const int xm1 = sidx[m0 + 64];
    const unsigned segkc = ((p & 1u) << 4) | ((p >> 1) << 3);   // {0,16,8,24}
    const unsigned lds_byte0 =
        ((((m0 >> 4) * 32 + ((p >> 1) << 4) + (m0 & 15u)) << 4) + ((p & 1u) << 3)) * 2u;
    // token m0+64 lands 4 M-tiles (4*32*16 elems = 4096 B) further
    // LDS byte base of Alds (low 32 bits of generic pointer == LDS offset)
    const unsigned ldsbase = (unsigned)(unsigned long long)(void*)&Alds[0][0];

    // --- compute identity: wave -> (M-tile pair, N-half)
    const unsigned lane = tid & 31u;
    const unsigned wave = tid >> 5;
    const unsigned mta  = (wave & 3u) * 2u;  // M-tiles mta, mta+1
    const unsigned nh   = wave >> 2;         // which 128-output half

    v8f acc[2][8];
    const v8f vzero = {0.f, 0.f, 0.f, 0.f, 0.f, 0.f, 0.f, 0.f};
#pragma unroll
    for (int mi = 0; mi < 2; ++mi)
#pragma unroll
        for (int j = 0; j < 8; ++j) acc[mi][j] = vzero;

    auto gsrc = [&](int kk, int xm) -> const unsigned short* {
        int tap = kk >> 3;
        int r = xm + tap - 4;
        r = r < 0 ? 0 : (r > INP_SIZE - 1 ? INP_SIZE - 1 : r);
        unsigned hi0 = ((unsigned)kk & 7u) << 5;
        return embbf + (size_t)r * HIDDEN + hi0 + segkc;
    };

    // async-stage chunk kk into buffer buf (per-lane 16B -> fragment slot)
    auto issue_async = [&](int kk, unsigned buf) {
        const unsigned short* g0 = gsrc(kk, xm0);
        const unsigned short* g1 = gsrc(kk, xm1);
        unsigned l0 = ldsbase + buf * 8192u + lds_byte0;
        unsigned l1 = l0 + 4096u;
        asm volatile("global_load_async_to_lds_b128 %0, %1, off"
                     :: "v"(l0), "v"(g0) : "memory");
        asm volatile("global_load_async_to_lds_b128 %0, %1, off"
                     :: "v"(l1), "v"(g1) : "memory");
    };

    issue_async(0, 0);

    for (int kk = 0; kk < NCHUNK; ++kk) {
        const unsigned buf = (unsigned)kk & 1u;
        // my async writes done -> then WG barrier: everyone's tile visible
        asm volatile("s_wait_asynccnt 0x0" ::: "memory");
        __syncthreads();

        if (kk + 1 < NCHUNK) {
            issue_async(kk + 1, buf ^ 1u);   // DMA next tile while we compute
            if (kk + 2 < NCHUNK) {           // warm L2 for the random gather
                __builtin_prefetch((const void*)gsrc(kk + 2, xm0), 0, 1);
                __builtin_prefetch((const void*)gsrc(kk + 2, xm1), 0, 1);
            }
        }

        // A fragments: contiguous 32B per lane from LDS
        const v16bf a0 = *(const v16bf*)(&Alds[buf][(mta * 32 + lane) << 4]);
        const v16bf a1 = *(const v16bf*)(&Alds[buf][((mta + 1) * 32 + lane) << 4]);

        // B fragments: contiguous 32B per lane from L2-resident swizzled table
        const unsigned short* bbase =
            wsB + (((size_t)kk * 16 + nh * 8) * 32 + lane) * 16;
#pragma unroll
        for (int j = 0; j < 8; ++j) {
            const v16bf b = *(const v16bf*)(bbase + (size_t)j * (32 * 16));
            acc[0][j] = __builtin_amdgcn_wmma_f32_16x16x32_bf16(
                false, a0, false, b, (short)0, acc[0][j], false, false);
            acc[1][j] = __builtin_amdgcn_wmma_f32_16x16x32_bf16(
                false, a1, false, b, (short)0, acc[1][j], false, false);
        }
    }

    // --- epilogue: C layout (ISA 7.12.2): VGPR r, lanes 0-15 -> M=r,
    //     lanes 16-31 -> M=r+8; N = lane&15 within the 16-wide tile.
    const unsigned col     = lane & 15u;
    const unsigned rowhalf = (lane >> 4) << 3;
#pragma unroll
    for (int mi = 0; mi < 2; ++mi) {
        const unsigned tokbase = wg * TOK_PER_WG + (mta + mi) * 16 + rowhalf;
#pragma unroll
        for (int j = 0; j < 8; ++j) {
            const unsigned o  = (nh * 8 + (unsigned)j) * 16 + col;
            const float    bo = bias[o];
#pragma unroll
            for (int r = 0; r < 8; ++r) {
                out[(size_t)(tokbase + r) * HIDDEN + o] = acc[mi][j][r] + bo;
            }
        }
    }
}

// ---------------------------------------------------------------------------
extern "C" void kernel_launch(void* const* d_in, const int* in_sizes, int n_in,
                              void* d_out, int out_size, void* d_ws, size_t ws_size,
                              hipStream_t stream) {
    const int*   x    = (const int*)d_in[0];
    const float* emb  = (const float*)d_in[1];
    const float* W    = (const float*)d_in[2];
    const float* bias = (const float*)d_in[3];
    float*       out  = (float*)d_out;

    unsigned short* ws_emb = (unsigned short*)d_ws;
    unsigned short* ws_B   = (unsigned short*)((char*)d_ws + WS_EMB_ELEMS * sizeof(unsigned short));

    const int n4 = (int)(WS_EMB_ELEMS / 4);
    prep_emb_bf16<<<(n4 + 255) / 256, 256, 0, stream>>>(emb, ws_emb);
    prep_w_swizzle<<<(int)(WS_B_ELEMS / 256), 256, 0, stream>>>(W, ws_B);
    conv_emb_wmma<<<NWG, 256, 0, stream>>>(x, ws_emb, ws_B, bias, out);
}
